// Model_90752658964498
// MI455X (gfx1250) — matrix operand order repaired and checked by reading, not compiled, not run on hardware
//
#include <hip/hip_runtime.h>

// ---------------------------------------------------------------------------
// GNN (GAT-style) 3-layer model for MI455X (gfx1250), wave32 + WMMA bf16.
// GEMMs -> v_wmma_f32_16x16x32_bf16. Gathers (x[src], x[dst]) and the
// segment_sum scatter are fused into the GEMM. A-panel staged to LDS 128-K
// wide per barrier round (4 WMMA steps / barrier pair) via the gfx1250
// async-to-LDS path. Block tile = 32x64 (2 M-subtiles share each B frag).
// ---------------------------------------------------------------------------

#define NN    20000
#define EE    100000
#define DD    768
#define HH    8
#define DHD   96
#define LL    3

typedef __bf16 bf16_t;
typedef bf16_t v16bf __attribute__((ext_vector_type(16)));
typedef float  v8f   __attribute__((ext_vector_type(8)));
typedef int    v4i_t __attribute__((vector_size(16)));   // matches async-LDS builtin param

union FragAB { uint4 q[2]; v16bf v; };

#if defined(__has_builtin)
#  if __has_builtin(__builtin_amdgcn_global_load_async_to_lds_b128)
#    define USE_ASYNC_LDS 1
#  endif
#endif
#ifndef USE_ASYNC_LDS
#  define USE_ASYNC_LDS 0
#endif

#define GLOBAL_AS __attribute__((address_space(1)))
#define LDS_AS    __attribute__((address_space(3)))

__device__ __forceinline__ float bf2f(unsigned short u) {
    union { unsigned int u; float f; } c; c.u = ((unsigned int)u) << 16; return c.f;
}
__device__ __forceinline__ unsigned short f2bf(float f) {
    union { float f; unsigned int u; } c; c.f = f;
    unsigned int r = c.u + 0x7FFFu + ((c.u >> 16) & 1u);   // round-to-nearest-even
    return (unsigned short)(r >> 16);
}
__device__ __forceinline__ unsigned int pack2bf(float lo, float hi) {
    return (unsigned int)f2bf(lo) | ((unsigned int)f2bf(hi) << 16);
}
__device__ __forceinline__ float dotw(unsigned int a, unsigned int b) {
    return bf2f((unsigned short)a) * bf2f((unsigned short)b) +
           bf2f((unsigned short)(a >> 16)) * bf2f((unsigned short)(b >> 16));
}

// --------------------------- converters ------------------------------------

// 8 elements per thread (f32 -> bf16)
__global__ void cvt_bf16_k(const float* __restrict__ in, unsigned short* __restrict__ out,
                           long long n8) {
    long long t = (long long)blockIdx.x * blockDim.x + threadIdx.x;
    if (t >= n8) return;
    long long base = t * 8;
    float4 a = *(const float4*)(in + base);
    float4 b = *(const float4*)(in + base + 4);
    uint4 o;
    o.x = pack2bf(a.x, a.y);
    o.y = pack2bf(a.z, a.w);
    o.z = pack2bf(b.x, b.y);
    o.w = pack2bf(b.z, b.w);
    *(uint4*)(out + base) = o;
}

// out[n*K + k] = bf16(in[k*Nc + n])   (weight transpose: Bt is [Nc, K])
__global__ void transpose_bf16_k(const float* __restrict__ in, unsigned short* __restrict__ out,
                                 int K, int Nc) {
    long long t = (long long)blockIdx.x * blockDim.x + threadIdx.x;
    if (t >= (long long)K * Nc) return;
    int n = (int)(t / K);
    int k = (int)(t % K);
    out[t] = f2bf(in[(long long)k * Nc + n]);
}

// --------------------------- WMMA GEMM -------------------------------------
// C[M, Dout] = act( gatherA @ B + bias ), optionally atomically scattered.
// A columns [0,K0) from A0 (rows via idx0 or identity), [K0,Ktot) from A1
// (rows via idx1 or identity). Bt is bf16 [Dout, Ktot] (pre-transposed).
// Block = 128 threads (4 waves); block tile = 32 rows x 64 cols.
// K staged to LDS 128 wide per barrier round => 4 wmma k-steps per 2 barriers.
__global__ __launch_bounds__(128)
void gemm_wmma_bf16(const unsigned short* __restrict__ A0, const long long* __restrict__ idx0, int K0,
                    const unsigned short* __restrict__ A1, const long long* __restrict__ idx1,
                    int Ktot,
                    const unsigned short* __restrict__ Bt,
                    const float* __restrict__ bias,
                    float* __restrict__ Cf32, unsigned short* __restrict__ Cbf16,
                    const long long* __restrict__ scat,
                    int M, int Dout, int act)
{
    __shared__ unsigned short Atile[32][128];   // 8 KB

    const int tid  = threadIdx.x;
    const int wave = tid >> 5;
    const int lane = tid & 31;
    const int kg   = lane >> 4;    // which k-half this lane holds
    const int ln   = lane & 15;

    const int rowBase = blockIdx.x * 32;
    const int col     = blockIdx.y * 64 + wave * 16 + ln;   // output column of this lane

    // staging map: thread t -> row sr = t>>2, 32 halves (64 B) at sc0 = (t&3)*32
    const int sr  = tid >> 2;
    const int sc0 = (tid & 3) * 32;
    long long row0 = rowBase + sr;
    long long row1 = row0;
    if (idx0) row0 = idx0[rowBase + sr];
    if (idx1) row1 = idx1[rowBase + sr];
    const int K1 = Ktot - K0;

    const v8f vzero = {0.f, 0.f, 0.f, 0.f, 0.f, 0.f, 0.f, 0.f};
    v8f acc[2]; acc[0] = vzero; acc[1] = vzero;

    for (int kb = 0; kb < Ktot; kb += 128) {
        // K0 is a multiple of 128 -> a staging round never straddles A0/A1
        const unsigned short* sp = (kb < K0)
            ? A0 + row0 * (long long)K0 + (kb + sc0)
            : A1 + row1 * (long long)K1 + (kb - K0 + sc0);
        unsigned short* lp = &Atile[sr][sc0];
#if USE_ASYNC_LDS
#pragma unroll
        for (int i = 0; i < 4; ++i)
            __builtin_amdgcn_global_load_async_to_lds_b128(
                (GLOBAL_AS v4i_t*)(sp + i * 8), (LDS_AS v4i_t*)(lp + i * 8), 0, 0);
        asm volatile("s_wait_asynccnt 0" ::: "memory");
#else
#pragma unroll
        for (int i = 0; i < 4; ++i)
            ((uint4*)lp)[i] = ((const uint4*)sp)[i];
#endif
        __syncthreads();

        const unsigned short* bp0 = Bt + (long long)col * Ktot + kb + kg * 8;
#pragma unroll
        for (int kk = 0; kk < 128; kk += 32) {
            // A frag (ISA 16-bit A 16x32 layout): lane ln, halves[0..7]=K kg*8..+7,
            // halves[8..15]=K 16+kg*8..+7 -> two contiguous 16B LDS reads
            FragAB a0, a1, b;
            a0.q[0] = *(const uint4*)&Atile[ln][kk + kg * 8];
            a0.q[1] = *(const uint4*)&Atile[ln][kk + 16 + kg * 8];
            a1.q[0] = *(const uint4*)&Atile[16 + ln][kk + kg * 8];
            a1.q[1] = *(const uint4*)&Atile[16 + ln][kk + 16 + kg * 8];
            // B frag: lane = column, halves[0..7]=K kg*8..+7, halves[8..15]=K 16+kg*8..+7
            b.q[0] = *(const uint4*)(bp0 + kk);
            b.q[1] = *(const uint4*)(bp0 + kk + 16);
            acc[0] = __builtin_amdgcn_wmma_f32_16x16x32_bf16(false, a0.v, false, b.v,
                                                             (short)0, acc[0], false, false);
            acc[1] = __builtin_amdgcn_wmma_f32_16x16x32_bf16(false, a1.v, false, b.v,
                                                             (short)0, acc[1], false, false);
        }
        __syncthreads();
    }

    const float bcol = bias ? bias[col] : 0.f;
#pragma unroll
    for (int mt = 0; mt < 2; ++mt) {
#pragma unroll
        for (int i = 0; i < 8; ++i) {
            int row = rowBase + mt * 16 + i + 8 * kg;   // C/D: VGPR i -> M = i + 8*(lane/16)
            float v = acc[mt][i] + bcol;
            if (act == 1) v = v > 0.f ? v : 0.2f * v;   // leaky_relu(0.2)
            if (scat) {
                long long drow = scat[row];
                atomicAdd(&Cf32[drow * (long long)Dout + col], v);
            } else {
                long long off = (long long)row * Dout + col;
                if (Cf32)  Cf32[off]  = v;
                if (Cbf16) Cbf16[off] = f2bf(v);
            }
        }
    }
}

// --------------------------- attention pieces ------------------------------

// aw[e,h] = leaky_relu( dot(qh, kh) / sqrt(D), 0.2 )   (12 x b128 per dot)
__global__ void qk_dot_k(const unsigned short* __restrict__ q, const unsigned short* __restrict__ k,
                         float* __restrict__ awl, int E) {
    int t = blockIdx.x * blockDim.x + threadIdx.x;
    if (t >= E * HH) return;
    int e = t >> 3, h = t & 7;
    long long base = (long long)e * DD + h * DHD;
    const uint4* qp = (const uint4*)(q + base);
    const uint4* kp = (const uint4*)(k + base);
    float s = 0.f;
#pragma unroll
    for (int i = 0; i < DHD / 8; ++i) {
        uint4 a = qp[i], b = kp[i];
        s += dotw(a.x, b.x) + dotw(a.y, b.y) + dotw(a.z, b.z) + dotw(a.w, b.w);
    }
    s *= 0.03608439182435161f;           // 1/sqrt(768)
    awl[t] = s > 0.f ? s : 0.2f * s;
}

// a[e,:] = exp(awl[e,:] @ Waw + baw); sc[dst[e],:] += a[e,:]
__global__ void attn_exp_scatter_k(const float* __restrict__ awl,
                                   const float* __restrict__ Waw, const float* __restrict__ baw,
                                   float* __restrict__ a, float* __restrict__ sc,
                                   const long long* __restrict__ dst, int E) {
    int e = blockIdx.x * blockDim.x + threadIdx.x;
    if (e >= E) return;
    float in[HH];
#pragma unroll
    for (int j = 0; j < HH; ++j) in[j] = awl[e * HH + j];
    long long drow = dst[e];
#pragma unroll
    for (int h = 0; h < HH; ++h) {
        float z = baw[h];
#pragma unroll
        for (int j = 0; j < HH; ++j) z += in[j] * Waw[j * HH + h];
        float av = __expf(z);
        a[e * HH + h] = av;
        atomicAdd(&sc[drow * HH + h], av);
    }
}

// out[e, j8..j8+7] = bf16( att * v ),  one b128 in / one b128 out per thread
__global__ void attv_k(const float* __restrict__ a, const float* __restrict__ sc,
                       const unsigned short* __restrict__ v, const long long* __restrict__ dst,
                       unsigned short* __restrict__ out, long long groups) {
    long long g = (long long)blockIdx.x * blockDim.x + threadIdx.x;
    if (g >= groups) return;                 // groups = E * (DD/8)
    long long e = g / (DD / 8);
    int j8 = (int)(g - e * (DD / 8)) * 8;    // 8-aligned, 96 % 8 == 0 -> single head
    int h = j8 / DHD;
    float denom = sc[dst[e] * HH + h];
    if (denom == 0.f) denom = 1.f;
    float att = a[e * HH + h] / denom;
    const uint4 vv = *(const uint4*)(v + e * DD + j8);
    uint4 o;
    o.x = pack2bf(att * bf2f((unsigned short)vv.x), att * bf2f((unsigned short)(vv.x >> 16)));
    o.y = pack2bf(att * bf2f((unsigned short)vv.y), att * bf2f((unsigned short)(vv.y >> 16)));
    o.z = pack2bf(att * bf2f((unsigned short)vv.z), att * bf2f((unsigned short)(vv.z >> 16)));
    o.w = pack2bf(att * bf2f((unsigned short)vv.w), att * bf2f((unsigned short)(vv.w >> 16)));
    *(uint4*)(out + e * DD + j8) = o;
}

// gated residual: z = t1 @ Wla2 + bla2 (D->2), la = softmax(z),
// x_out = x_in*la0 + h*la1.  One wave (32 lanes) per node.
__global__ __launch_bounds__(128)
void combine_k(const unsigned short* __restrict__ t1,
               const float* __restrict__ Wla2, const float* __restrict__ bla2,
               const float* __restrict__ x_in, const float* __restrict__ h,
               float* __restrict__ x_out, int N) {
    int node = blockIdx.x * (blockDim.x >> 5) + (threadIdx.x >> 5);
    int lane = threadIdx.x & 31;
    if (node >= N) return;
    float z0 = 0.f, z1 = 0.f;
    for (int d = lane; d < DD; d += 32) {
        float tv = bf2f(t1[(long long)node * DD + d]);
        z0 += tv * Wla2[d * 2 + 0];
        z1 += tv * Wla2[d * 2 + 1];
    }
#pragma unroll
    for (int o = 16; o > 0; o >>= 1) {
        z0 += __shfl_xor(z0, o, 32);
        z1 += __shfl_xor(z1, o, 32);
    }
    z0 += bla2[0]; z1 += bla2[1];
    float m  = fmaxf(z0, z1);
    float e0 = __expf(z0 - m), e1 = __expf(z1 - m);
    float inv = 1.f / (e0 + e1);
    float la0 = e0 * inv, la1 = e1 * inv;
    for (int d = lane; d < DD; d += 32) {
        long long off = (long long)node * DD + d;
        x_out[off] = x_in[off] * la0 + h[off] * la1;   // in-place safe per-thread
    }
}

// --------------------------- host launcher ---------------------------------

static inline int cdiv(long long a, int b) { return (int)((a + b - 1) / b); }

extern "C" void kernel_launch(void* const* d_in, const int* in_sizes, int n_in,
                              void* d_out, int out_size, void* d_ws, size_t ws_size,
                              hipStream_t stream)
{
    (void)in_sizes; (void)n_in; (void)out_size; (void)ws_size;

    const float* x0   = (const float*)d_in[0];
    const float* ea   = (const float*)d_in[1];
    const float* Wck  = (const float*)d_in[2];
    const float* bck  = (const float*)d_in[3];
    const float* Wq   = (const float*)d_in[4];
    const float* bq   = (const float*)d_in[5];
    const float* Wv   = (const float*)d_in[6];
    const float* bv   = (const float*)d_in[7];
    const float* Waw  = (const float*)d_in[8];
    const float* baw  = (const float*)d_in[9];
    const float* Wc   = (const float*)d_in[10];
    const float* bc   = (const float*)d_in[11];
    const float* Wlx  = (const float*)d_in[12];
    const float* blx  = (const float*)d_in[13];
    const float* Wla1 = (const float*)d_in[14];
    const float* bla1 = (const float*)d_in[15];
    const float* Wla2 = (const float*)d_in[16];
    const float* bla2 = (const float*)d_in[17];
    const long long* ei  = (const long long*)d_in[18];
    const long long* src = ei;
    const long long* dst = ei + EE;
    float* out = (float*)d_out;

    // bump allocator over workspace
    char* p = (char*)d_ws;
    auto alloc = [&](size_t bytes) -> void* {
        void* r = (void*)p;
        p += (bytes + 255) & ~(size_t)255;
        return r;
    };

    unsigned short *wt_q[LL], *wt_ck[LL], *wt_v[LL], *wt_c[LL], *wt_lx[LL], *wt_la1[LL];
    for (int l = 0; l < LL; ++l) {
        wt_q[l]   = (unsigned short*)alloc((size_t)DD * DD * 2);
        wt_ck[l]  = (unsigned short*)alloc((size_t)DD * 2 * DD * 2);
        wt_v[l]   = (unsigned short*)alloc((size_t)DD * DD * 2);
        wt_c[l]   = (unsigned short*)alloc((size_t)DD * DD * 2);
        wt_lx[l]  = (unsigned short*)alloc((size_t)DD * DD * 2);
        wt_la1[l] = (unsigned short*)alloc((size_t)DD * 2 * DD * 2);
    }
    unsigned short* x_bf   = (unsigned short*)alloc((size_t)NN * DD * 2);
    unsigned short* ea_bf  = (unsigned short*)alloc((size_t)EE * DD * 2);
    unsigned short* q_bf   = (unsigned short*)alloc((size_t)EE * DD * 2);  // reused as att*v
    unsigned short* k_bf   = (unsigned short*)alloc((size_t)EE * DD * 2);  // reused as t1
    unsigned short* v_bf   = (unsigned short*)alloc((size_t)EE * DD * 2);
    float* awl    = (float*)alloc((size_t)EE * HH * 4);
    float* a_buf  = (float*)alloc((size_t)EE * HH * 4);
    float* sc     = (float*)alloc((size_t)NN * HH * 4);
    float* h_buf  = (float*)alloc((size_t)NN * DD * 4);
    unsigned short* h_bf  = (unsigned short*)alloc((size_t)NN * DD * 2);
    unsigned short* lx_bf = (unsigned short*)alloc((size_t)NN * DD * 2);
    float* x_buf  = (float*)alloc((size_t)NN * DD * 4);

    // ---- one-time weight transpose/convert + edge_attr convert ----
    for (int l = 0; l < LL; ++l) {
        transpose_bf16_k<<<cdiv((long long)DD * DD, 256), 256, 0, stream>>>(
            Wq + (size_t)l * DD * DD, wt_q[l], DD, DD);
        transpose_bf16_k<<<cdiv((long long)2 * DD * DD, 256), 256, 0, stream>>>(
            Wck + (size_t)l * 2 * DD * DD, wt_ck[l], 2 * DD, DD);
        transpose_bf16_k<<<cdiv((long long)DD * DD, 256), 256, 0, stream>>>(
            Wv + (size_t)l * DD * DD, wt_v[l], DD, DD);
        transpose_bf16_k<<<cdiv((long long)DD * DD, 256), 256, 0, stream>>>(
            Wc + (size_t)l * DD * DD, wt_c[l], DD, DD);
        transpose_bf16_k<<<cdiv((long long)DD * DD, 256), 256, 0, stream>>>(
            Wlx + (size_t)l * DD * DD, wt_lx[l], DD, DD);
        transpose_bf16_k<<<cdiv((long long)2 * DD * DD, 256), 256, 0, stream>>>(
            Wla1 + (size_t)l * 2 * DD * DD, wt_la1[l], 2 * DD, DD);
    }
    cvt_bf16_k<<<cdiv((long long)EE * DD / 8, 256), 256, 0, stream>>>(
        ea, ea_bf, (long long)EE * DD / 8);

    const dim3 gE(EE / 32, DD / 64);
    const dim3 gN(NN / 32, DD / 64);

    for (int l = 0; l < LL; ++l) {
        const float* xs = (l == 0) ? x0 : x_buf;
        float*       xd = (l == LL - 1) ? out : x_buf;

        cvt_bf16_k<<<cdiv((long long)NN * DD / 8, 256), 256, 0, stream>>>(
            xs, x_bf, (long long)NN * DD / 8);

        // q = x[dst] @ Wq + bq
        gemm_wmma_bf16<<<gE, 128, 0, stream>>>(x_bf, dst, DD, nullptr, nullptr, DD,
                                               wt_q[l], bq + (size_t)l * DD,
                                               nullptr, q_bf, nullptr, EE, DD, 0);
        // k = [edge_attr || x[src]] @ Wck + bck
        gemm_wmma_bf16<<<gE, 128, 0, stream>>>(ea_bf, nullptr, DD, x_bf, src, 2 * DD,
                                               wt_ck[l], bck + (size_t)l * DD,
                                               nullptr, k_bf, nullptr, EE, DD, 0);
        // v = x[src] @ Wv + bv
        gemm_wmma_bf16<<<gE, 128, 0, stream>>>(x_bf, src, DD, nullptr, nullptr, DD,
                                               wt_v[l], bv + (size_t)l * DD,
                                               nullptr, v_bf, nullptr, EE, DD, 0);

        // per-head attention logits + GAT softmax accumulation
        qk_dot_k<<<cdiv((long long)EE * HH, 256), 256, 0, stream>>>(q_bf, k_bf, awl, EE);
        (void)hipMemsetAsync(sc, 0, (size_t)NN * HH * 4, stream);
        attn_exp_scatter_k<<<cdiv(EE, 256), 256, 0, stream>>>(
            awl, Waw + (size_t)l * HH * HH, baw + (size_t)l * HH, a_buf, sc, dst, EE);

        // att * v  (overwrites q buffer)
        attv_k<<<cdiv((long long)EE * (DD / 8), 256), 256, 0, stream>>>(
            a_buf, sc, v_bf, dst, q_bf, (long long)EE * (DD / 8));

        // h = segment_sum( (att*v) @ Wc + bc , dst )  -- fused atomic scatter
        (void)hipMemsetAsync(h_buf, 0, (size_t)NN * DD * 4, stream);
        gemm_wmma_bf16<<<gE, 128, 0, stream>>>(q_bf, nullptr, DD, nullptr, nullptr, DD,
                                               wt_c[l], bc + (size_t)l * DD,
                                               h_buf, nullptr, dst, EE, DD, 0);

        // lx = x @ Wlx + blx
        gemm_wmma_bf16<<<gN, 128, 0, stream>>>(x_bf, nullptr, DD, nullptr, nullptr, DD,
                                               wt_lx[l], blx + (size_t)l * DD,
                                               nullptr, lx_bf, nullptr, NN, DD, 0);

        // t1 = leaky_relu([lx || h] @ Wla1 + bla1)   (t1 reuses k buffer)
        cvt_bf16_k<<<cdiv((long long)NN * DD / 8, 256), 256, 0, stream>>>(
            h_buf, h_bf, (long long)NN * DD / 8);
        gemm_wmma_bf16<<<gN, 128, 0, stream>>>(lx_bf, nullptr, DD, h_bf, nullptr, 2 * DD,
                                               wt_la1[l], bla1 + (size_t)l * DD,
                                               nullptr, k_bf, nullptr, NN, DD, 1);

        // gated residual combine
        combine_k<<<cdiv(NN, 4), 128, 0, stream>>>(
            k_bf, Wla2 + (size_t)l * DD * 2, bla2 + (size_t)l * 2, xs, h_buf, xd, NN);
    }
}
